// AutoGNNIDS_83803401879713
// MI455X (gfx1250) — matrix operand order repaired and checked by reading, not compiled, not run on hardware
//
#include <hip/hip_runtime.h>
#include <math.h>

// GNN anomaly-detection pipeline for MI455X (gfx1250, wave32).
// Dense layers -> v_wmma_f32_16x16x32_f16 (fp32 in -> f16 convert, fp32 acc).
// All GEMM shapes satisfy M%16==0, K%32==0, N%16==0 -> unconditional vector loads.
// bias/addIn/activation are template params so the epilogue has no per-element
// pointer-null branches (previously caused 8 serialized load->wait pairs/tile).
// Graph segment ops -> VALU + float atomics (L2-resident: 192MB L2 >> 26MB h-matrix).

typedef __attribute__((ext_vector_type(16))) _Float16 v16h;
typedef __attribute__((ext_vector_type(8)))  float    v8f;

template<int ACT>
static __device__ __forceinline__ float actf(float v) {
  if (ACT == 1) return v > 0.f ? v : 0.f;                 // relu
  if (ACT == 2) return v > 0.f ? v : (__expf(v) - 1.f);   // elu
  return v;
}
static __device__ __forceinline__ float sigm(float x) { return 1.f / (1.f + __expf(-x)); }

// ---------------- WMMA GEMM: out = act(A[MxK] @ B[KxN] + bias + addIn) ----------------
// Requires: M%16==0, K%32==0, Ncols%16==0 (true for every layer here).
// One wave computes one 16x16 output tile; K-chunks of 32.
// A layout (ISA 7.12.2, 16-bit A 16x32): lane l holds row m=l&15; halves are two
//   contiguous 8-float runs at k0+laneHi*8 and k0+16+laneHi*8  -> 4x b128 loads.
// B layout (32x16): lane l holds col n=l&15; half i -> k = k0+laneHi*8+i (i<8), k0+16+laneHi*8+(i-8) (i>=8).
// C/D layout: lane l, float j -> row = j + laneHi*8, col = l&15.
template<int ACT, bool HAS_BIAS, bool HAS_ADD>
__global__ void gemm_wmma_k(const float* __restrict__ A, const float* __restrict__ B,
                            const float* __restrict__ bias, const float* __restrict__ addIn,
                            float* __restrict__ out, int M, int K, int Ncols)
{
  const int lane   = threadIdx.x & 31;
  const int waveId = threadIdx.x >> 5;
  const int nTilesN = Ncols >> 4;
  const int tile = blockIdx.x * 8 + waveId;            // wave-uniform
  if (tile >= (M >> 4) * nTilesN) return;              // uniform exit: EXEC stays all-ones
  const int rowTile = tile / nTilesN;
  const int colTile = tile - rowTile * nTilesN;
  const int laneHi  = lane >> 4;
  const int mloc    = lane & 15;

  const float* Arow = A + (long)(rowTile * 16 + mloc) * K + laneHi * 8;
  const float* Bcol = B + colTile * 16 + mloc + (long)(laneHi * 8) * Ncols;

  v8f acc = {};
  for (int k0 = 0; k0 < K; k0 += 32) {
    const float4 a0 = *(const float4*)(Arow + k0);
    const float4 a1 = *(const float4*)(Arow + k0 + 4);
    const float4 a2 = *(const float4*)(Arow + k0 + 16);
    const float4 a3 = *(const float4*)(Arow + k0 + 20);
    v16h a;
    a[0] = (_Float16)a0.x;  a[1] = (_Float16)a0.y;  a[2]  = (_Float16)a0.z;  a[3]  = (_Float16)a0.w;
    a[4] = (_Float16)a1.x;  a[5] = (_Float16)a1.y;  a[6]  = (_Float16)a1.z;  a[7]  = (_Float16)a1.w;
    a[8] = (_Float16)a2.x;  a[9] = (_Float16)a2.y;  a[10] = (_Float16)a2.z;  a[11] = (_Float16)a2.w;
    a[12] = (_Float16)a3.x; a[13] = (_Float16)a3.y; a[14] = (_Float16)a3.z;  a[15] = (_Float16)a3.w;
    v16h b;
#pragma unroll
    for (int i = 0; i < 16; ++i)
      b[i] = (_Float16)Bcol[(long)(k0 + i + (i >= 8 ? 8 : 0)) * Ncols];
    acc = __builtin_amdgcn_wmma_f32_16x16x32_f16(false, a, false, b, (short)0, acc, false, false);
  }

  const int  col  = colTile * 16 + mloc;
  const long rb   = (long)(rowTile * 16 + laneHi * 8) * Ncols + col;
  const float bval = HAS_BIAS ? bias[col] : 0.f;

  float addv[8];
  if (HAS_ADD) {
#pragma unroll
    for (int j = 0; j < 8; ++j) addv[j] = addIn[rb + (long)j * Ncols];  // batched, no branches
  }
#pragma unroll
  for (int j = 0; j < 8; ++j) {
    float v = acc[j] + bval;
    if (HAS_ADD) v += addv[j];
    out[rb + (long)j * Ncols] = actf<ACT>(v);
  }
}

// ---------------- elementwise / graph kernels ----------------
__global__ void fill_k(float* p, float v, long n) {
  for (long i = blockIdx.x * (long)blockDim.x + threadIdx.x; i < n; i += (long)gridDim.x * blockDim.x)
    p[i] = v;
}

// in-degree + edge_attr sum per destination
__global__ void deg_ea_k(const int* __restrict__ src0, const int* __restrict__ dst0,
                         const float* __restrict__ edge_attr, float* deg, float* ea_sum, int E)
{
  for (long e = blockIdx.x * (long)blockDim.x + threadIdx.x; e < E; e += (long)gridDim.x * blockDim.x) {
    int d = dst0[e];
    atomicAdd(&deg[d], 1.f);
#pragma unroll
    for (int f = 0; f < 16; ++f) atomicAdd(&ea_sum[(long)d * 16 + f], edge_attr[e * 16 + f]);
  }
}

__global__ void divdeg_k(float* buf, const float* __restrict__ deg, int Nn, int C) {
  long total = (long)Nn * C;
  for (long i = blockIdx.x * (long)blockDim.x + threadIdx.x; i < total; i += (long)gridDim.x * blockDim.x) {
    float d = deg[i / C];
    buf[i] /= (d < 1.f ? 1.f : d);
  }
}

// asrc/adst per (node, head)
__global__ void node_att_k(const float* __restrict__ hmat, const float* __restrict__ a_s,
                           const float* __restrict__ a_d, float* asrc, float* adst,
                           int Nn, int H, int C)
{
  long total = (long)Nn * H;
  for (long i = blockIdx.x * (long)blockDim.x + threadIdx.x; i < total; i += (long)gridDim.x * blockDim.x) {
    int h = (int)(i % H);
    long n = i / H;
    const float* hp = hmat + (n * H + h) * C;
    float ss = 0.f, sd = 0.f;
    for (int c = 0; c < C; ++c) { ss += hp[c] * a_s[h * C + c]; sd += hp[c] * a_d[h * C + c]; }
    asrc[i] = ss; adst[i] = sd;
  }
}

// v[h,f] = sum_c We[f, h*C+c] * ae[h,c]   (pre-contract edge attention)
__global__ void att_vec_k(const float* __restrict__ We, const float* __restrict__ ae,
                          float* v, int H, int C)
{
  int idx = blockIdx.x * blockDim.x + threadIdx.x;
  if (idx >= H * 16) return;
  int h = idx / 16, f = idx % 16;
  float s = 0.f;
  for (int c = 0; c < C; ++c) s += We[(long)f * H * C + h * C + c] * ae[h * C + c];
  v[idx] = s;
}

static __device__ __forceinline__ unsigned enc_f(float f) {
  unsigned u = __float_as_uint(f);
  return (u & 0x80000000u) ? ~u : (u | 0x80000000u);
}
static __device__ __forceinline__ float dec_f(unsigned u) {
  return __uint_as_float((u & 0x80000000u) ? (u & 0x7FFFFFFFu) : ~u);
}

// leaky-relu logits + segment-max (encoded uint atomicMax); edges [0,E) real, [E,E+Nn) self-loops
__global__ void edge_logit_max_k(const int* __restrict__ src0, const int* __restrict__ dst0,
                                 const float* __restrict__ edge_attr, const float* __restrict__ ea_mean,
                                 const float* __restrict__ asrc, const float* __restrict__ adst,
                                 const float* __restrict__ vatt, float* logit, unsigned* maxenc,
                                 int E, int Nn, int H)
{
  long total = (long)(E + Nn) * H;
  for (long idx = blockIdx.x * (long)blockDim.x + threadIdx.x; idx < total; idx += (long)gridDim.x * blockDim.x) {
    int h = (int)(idx % H);
    long e = idx / H;
    int s, d; const float* ea;
    if (e < E) { s = src0[e]; d = dst0[e]; ea = edge_attr + e * 16; }
    else       { s = d = (int)(e - E);     ea = ea_mean + (e - E) * 16; }
    float eatt = 0.f;
    const float* v = vatt + h * 16;
#pragma unroll
    for (int f = 0; f < 16; ++f) eatt += ea[f] * v[f];
    float x = asrc[(long)s * H + h] + adst[(long)d * H + h] + eatt;
    x = x > 0.f ? x : 0.2f * x;
    logit[idx] = x;
    atomicMax(&maxenc[(long)d * H + h], enc_f(x));
  }
}

__global__ void edge_exp_k(const int* __restrict__ dst0, const float* __restrict__ logit,
                           const unsigned* __restrict__ maxenc, float* expv, float* denom,
                           int E, int Nn, int H)
{
  long total = (long)(E + Nn) * H;
  for (long idx = blockIdx.x * (long)blockDim.x + threadIdx.x; idx < total; idx += (long)gridDim.x * blockDim.x) {
    int h = (int)(idx % H);
    long e = idx / H;
    int d = (e < E) ? dst0[e] : (int)(e - E);
    float m = dec_f(maxenc[(long)d * H + h]);
    float ex = __expf(logit[idx] - m);
    expv[idx] = ex;
    atomicAdd(&denom[(long)d * H + h], ex);
  }
}

// out[d,h,c] += (expv/denom[d,h]) * h[s,h,c]
__global__ void gat_aggr_k(const int* __restrict__ src0, const int* __restrict__ dst0,
                           const float* __restrict__ hmat, const float* __restrict__ expv,
                           const float* __restrict__ denom, float* out, int E, int Nn, int H, int C)
{
  long total = (long)(E + Nn) * H * C;
  for (long idx = blockIdx.x * (long)blockDim.x + threadIdx.x; idx < total; idx += (long)gridDim.x * blockDim.x) {
    int c = (int)(idx % C);
    long t = idx / C;
    int h = (int)(t % H);
    long e = t / H;
    int s, d;
    if (e < E) { s = src0[e]; d = dst0[e]; } else { s = d = (int)(e - E); }
    float alpha = expv[e * H + h] / (denom[(long)d * H + h] + 1e-16f);
    atomicAdd(&out[((long)d * H + h) * C + c], alpha * hmat[((long)s * H + h) * C + c]);
  }
}

__global__ void bias_act_elu_k(float* buf, const float* __restrict__ bias, long total, int C) {
  for (long i = blockIdx.x * (long)blockDim.x + threadIdx.x; i < total; i += (long)gridDim.x * blockDim.x)
    buf[i] = actf<2>(buf[i] + bias[i % C]);
}

__global__ void alpha_out_k(const int* __restrict__ dst0, const float* __restrict__ expv,
                            const float* __restrict__ denom, float* out, int E, int Nn)
{
  long total = (long)E + Nn;
  for (long e = blockIdx.x * (long)blockDim.x + threadIdx.x; e < total; e += (long)gridDim.x * blockDim.x) {
    int d = (e < E) ? dst0[e] : (int)(e - E);
    out[e] = expv[e] / (denom[d] + 1e-16f);
  }
}

__global__ void sage_aggr_k(const int* __restrict__ src0, const int* __restrict__ dst0,
                            const float* __restrict__ hin, float* aggr, int E, int F)
{
  long total = (long)E * F;
  for (long idx = blockIdx.x * (long)blockDim.x + threadIdx.x; idx < total; idx += (long)gridDim.x * blockDim.x) {
    int f = (int)(idx % F);
    long e = idx / F;
    atomicAdd(&aggr[(long)dst0[e] * F + f], hin[(long)src0[e] * F + f]);
  }
}

__global__ void concat_k(const float* __restrict__ a, const float* __restrict__ b,
                         float* out, int Nn, int Ca, int Cb)
{
  int Ct = Ca + Cb;
  long total = (long)Nn * Ct;
  for (long i = blockIdx.x * (long)blockDim.x + threadIdx.x; i < total; i += (long)gridDim.x * blockDim.x) {
    long n = i / Ct; int j = (int)(i % Ct);
    out[i] = (j < Ca) ? a[n * Ca + j] : b[n * Cb + (j - Ca)];
  }
}

__global__ void adj_recon_k(const int* __restrict__ src0, const int* __restrict__ dst0,
                            const float* __restrict__ z, float* out, int E)
{
  for (long e = blockIdx.x * (long)blockDim.x + threadIdx.x; e < E; e += (long)gridDim.x * blockDim.x) {
    const float* zs = z + (long)src0[e] * 96;
    const float* zd = z + (long)dst0[e] * 96;
    float s = 0.f;
#pragma unroll
    for (int c = 0; c < 96; ++c) s += zs[c] * zd[c];
    out[e] = sigm(s);
  }
}

__global__ void score_k(const float* __restrict__ hid, const float* __restrict__ Wa2,
                        const float* __restrict__ ba2, float* out, int Nn)
{
  for (long n = blockIdx.x * (long)blockDim.x + threadIdx.x; n < Nn; n += (long)gridDim.x * blockDim.x) {
    float s = ba2[0];
#pragma unroll
    for (int c = 0; c < 32; ++c) s += hid[n * 32 + c] * Wa2[c];
    out[n] = sigm(s);
  }
}

// ---------------- host launcher ----------------
extern "C" void kernel_launch(void* const* d_in, const int* in_sizes, int n_in,
                              void* d_out, int out_size, void* d_ws, size_t ws_size,
                              hipStream_t stream)
{
  const int Nn = in_sizes[0] / 64;      // 50000
  const int E  = in_sizes[1] / 2;      // 800000
  const int EN = E + Nn;

  const float* x    = (const float*)d_in[0];
  const int*   ei   = (const int*)d_in[1];
  const int*   src0 = ei;
  const int*   dst0 = ei + E;
  const float* edge_attr = (const float*)d_in[2];
  const float* W1  = (const float*)d_in[3];
  const float* as1 = (const float*)d_in[4];
  const float* ad1 = (const float*)d_in[5];
  const float* We1 = (const float*)d_in[6];
  const float* ae1 = (const float*)d_in[7];
  const float* b1  = (const float*)d_in[8];
  const float* W2  = (const float*)d_in[9];
  const float* as2 = (const float*)d_in[10];
  const float* ad2 = (const float*)d_in[11];
  const float* We2 = (const float*)d_in[12];
  const float* ae2 = (const float*)d_in[13];
  const float* b2  = (const float*)d_in[14];
  const float* Wl1 = (const float*)d_in[15];
  const float* bl1 = (const float*)d_in[16];
  const float* Wr1 = (const float*)d_in[17];
  const float* Wl2 = (const float*)d_in[18];
  const float* bl2 = (const float*)d_in[19];
  const float* Wr2 = (const float*)d_in[20];
  const float* Wf  = (const float*)d_in[21];
  const float* bf  = (const float*)d_in[22];
  const float* Wd1 = (const float*)d_in[23];
  const float* bd1 = (const float*)d_in[24];
  const float* Wd2 = (const float*)d_in[25];
  const float* bd2 = (const float*)d_in[26];
  const float* Wa1 = (const float*)d_in[27];
  const float* ba1 = (const float*)d_in[28];
  const float* Wa2 = (const float*)d_in[29];
  const float* ba2 = (const float*)d_in[30];

  // output slots (return order: z, scores, adj_recon, feat_recon, alpha2)
  float* outp       = (float*)d_out;
  float* out_z      = outp;
  float* out_scores = out_z + (long)Nn * 96;
  float* out_adj    = out_scores + Nn;
  float* out_feat   = out_adj + E;
  float* out_alpha2 = out_feat + (long)Nn * 64;

  // workspace arena
  float* wsp = (float*)d_ws;
  size_t off = 0;
  auto alloc = [&](size_t n) { float* p = wsp + off; off += n; return p; };
  float*    deg     = alloc(Nn);
  float*    ea_mean = alloc((size_t)Nn * 16);     // holds sum, then mean in place
  float*    hmat1   = alloc((size_t)Nn * 128);
  float*    asrc1   = alloc((size_t)Nn * 4);
  float*    adst1   = alloc((size_t)Nn * 4);
  float*    vatt1   = alloc(64);
  float*    logit1  = alloc((size_t)EN * 4);
  float*    expv1   = alloc((size_t)EN * 4);
  unsigned* maxenc1 = (unsigned*)alloc((size_t)Nn * 4);
  float*    denom1  = alloc((size_t)Nn * 4);
  float*    agg1    = alloc((size_t)Nn * 128);    // becomes h1 in place
  float*    hmat2   = alloc((size_t)Nn * 32);
  float*    asrc2   = alloc(Nn);
  float*    adst2   = alloc(Nn);
  float*    vatt2   = alloc(16);
  float*    logit2  = alloc(EN);
  float*    expv2   = alloc(EN);
  unsigned* maxenc2 = (unsigned*)alloc(Nn);
  float*    denom2  = alloc(Nn);
  float*    agg2    = alloc((size_t)Nn * 32);     // becomes z_gat in place
  float*    sagg1   = alloc((size_t)Nn * 64);
  float*    stmp    = alloc((size_t)Nn * 64);
  float*    sbuf    = alloc((size_t)Nn * 64);     // s
  float*    sagg2   = alloc((size_t)Nn * 64);
  float*    zsage   = alloc((size_t)Nn * 64);
  float*    zcat    = alloc((size_t)Nn * 96);
  float*    hid     = alloc((size_t)Nn * 32);
  (void)ws_size; (void)n_in; (void)out_size;

  auto blocks = [](long n) { return (unsigned)((n + 255) / 256); };
  auto fill = [&](float* p, float v, long n) { fill_k<<<blocks(n), 256, 0, stream>>>(p, v, n); };

  // static-dispatch GEMM: template on (act, hasBias, hasAdd)
  auto gemm = [&](const float* A, const float* B, const float* bias, const float* addIn,
                  float* out, int M, int K, int Ncols, int act) {
    unsigned g = (unsigned)(((long)(M / 16) * (Ncols / 16) + 7) / 8);
#define GEMM_CASE(ACTV, BV, AV)                                                   \
    gemm_wmma_k<ACTV, BV, AV><<<g, 256, 0, stream>>>(A, B, bias, addIn, out, M, K, Ncols)
#define GEMM_ACT(ACTV)                                                            \
    { if (bias) { if (addIn) GEMM_CASE(ACTV, true,  true);                        \
                  else       GEMM_CASE(ACTV, true,  false); }                     \
      else      { if (addIn) GEMM_CASE(ACTV, false, true);                        \
                  else       GEMM_CASE(ACTV, false, false); } }
    if (act == 1)      GEMM_ACT(1)
    else if (act == 2) GEMM_ACT(2)
    else               GEMM_ACT(0)
#undef GEMM_ACT
#undef GEMM_CASE
  };

  // zero/neutral init of all accumulators (graph-replay safe)
  fill(deg, 0.f, Nn);
  fill(ea_mean, 0.f, (long)Nn * 16);
  fill((float*)maxenc1, 0.f, (long)Nn * 4);   // encoded -inf == 0
  fill(denom1, 0.f, (long)Nn * 4);
  fill(agg1, 0.f, (long)Nn * 128);
  fill((float*)maxenc2, 0.f, Nn);
  fill(denom2, 0.f, Nn);
  fill(agg2, 0.f, (long)Nn * 32);
  fill(sagg1, 0.f, (long)Nn * 64);
  fill(sagg2, 0.f, (long)Nn * 64);

  // degrees + self-loop edge-attr mean
  deg_ea_k<<<blocks(E), 256, 0, stream>>>(src0, dst0, edge_attr, deg, ea_mean, E);
  divdeg_k<<<blocks((long)Nn * 16), 256, 0, stream>>>(ea_mean, deg, Nn, 16);

  // ---- GAT layer 1 (H=4, C=32) ----
  gemm(x, W1, nullptr, nullptr, hmat1, Nn, 64, 128, 0);
  node_att_k<<<blocks((long)Nn * 4), 256, 0, stream>>>(hmat1, as1, ad1, asrc1, adst1, Nn, 4, 32);
  att_vec_k<<<1, 64, 0, stream>>>(We1, ae1, vatt1, 4, 32);
  edge_logit_max_k<<<blocks((long)EN * 4), 256, 0, stream>>>(src0, dst0, edge_attr, ea_mean,
                                                             asrc1, adst1, vatt1, logit1, maxenc1, E, Nn, 4);
  edge_exp_k<<<blocks((long)EN * 4), 256, 0, stream>>>(dst0, logit1, maxenc1, expv1, denom1, E, Nn, 4);
  gat_aggr_k<<<blocks((long)EN * 128), 256, 0, stream>>>(src0, dst0, hmat1, expv1, denom1, agg1, E, Nn, 4, 32);
  bias_act_elu_k<<<blocks((long)Nn * 128), 256, 0, stream>>>(agg1, b1, (long)Nn * 128, 128); // h1 = elu

  // ---- GAT layer 2 (H=1, C=32) ----
  gemm(agg1, W2, nullptr, nullptr, hmat2, Nn, 128, 32, 0);
  node_att_k<<<blocks(Nn), 256, 0, stream>>>(hmat2, as2, ad2, asrc2, adst2, Nn, 1, 32);
  att_vec_k<<<1, 64, 0, stream>>>(We2, ae2, vatt2, 1, 32);
  edge_logit_max_k<<<blocks(EN), 256, 0, stream>>>(src0, dst0, edge_attr, ea_mean,
                                                   asrc2, adst2, vatt2, logit2, maxenc2, E, Nn, 1);
  edge_exp_k<<<blocks(EN), 256, 0, stream>>>(dst0, logit2, maxenc2, expv2, denom2, E, Nn, 1);
  gat_aggr_k<<<blocks((long)EN * 32), 256, 0, stream>>>(src0, dst0, hmat2, expv2, denom2, agg2, E, Nn, 1, 32);
  bias_act_elu_k<<<blocks((long)Nn * 32), 256, 0, stream>>>(agg2, b2, (long)Nn * 32, 32);    // z_gat = elu
  alpha_out_k<<<blocks(EN), 256, 0, stream>>>(dst0, expv2, denom2, out_alpha2, E, Nn);

  // ---- SAGE layers (mean aggr, F=64) ----
  sage_aggr_k<<<blocks((long)E * 64), 256, 0, stream>>>(src0, dst0, x, sagg1, E, 64);
  divdeg_k<<<blocks((long)Nn * 64), 256, 0, stream>>>(sagg1, deg, Nn, 64);
  gemm(sagg1, Wl1, bl1, nullptr, stmp, Nn, 64, 64, 0);
  gemm(x, Wr1, nullptr, stmp, sbuf, Nn, 64, 64, 1);          // s = relu
  sage_aggr_k<<<blocks((long)E * 64), 256, 0, stream>>>(src0, dst0, sbuf, sagg2, E, 64);
  divdeg_k<<<blocks((long)Nn * 64), 256, 0, stream>>>(sagg2, deg, Nn, 64);
  gemm(sagg2, Wl2, bl2, nullptr, stmp, Nn, 64, 64, 0);
  gemm(sbuf, Wr2, nullptr, stmp, zsage, Nn, 64, 64, 1);      // z_sage = relu

  // ---- fusion + heads ----
  concat_k<<<blocks((long)Nn * 96), 256, 0, stream>>>(agg2, zsage, zcat, Nn, 32, 64);
  gemm(zcat, Wf, bf, nullptr, out_z, Nn, 96, 96, 1);         // z = relu -> directly to d_out
  adj_recon_k<<<blocks(E), 256, 0, stream>>>(src0, dst0, out_z, out_adj, E);
  gemm(out_z, Wd1, bd1, nullptr, hid, Nn, 96, 32, 1);
  gemm(hid, Wd2, bd2, nullptr, out_feat, Nn, 32, 64, 0);     // feat_recon
  gemm(out_z, Wa1, ba1, nullptr, hid, Nn, 96, 32, 1);
  score_k<<<blocks(Nn), 256, 0, stream>>>(hid, Wa2, ba2, out_scores, Nn);
}
